// MoELayer_67834713473243
// MI455X (gfx1250) — compile-verified, matrix operand order checked by reading, not run on hardware
//
#include <hip/hip_runtime.h>
#include <hip/hip_bf16.h>

// ---------------------------------------------------------------------------
// MoE (top-2 of 8 experts), T=8192 tokens, H=1024, F=2048.
// bf16 WMMA (v_wmma_f32_16x16x32_bf16) with fp32 accumulate; weights packed
// once per call into per-lane WMMA B-fragment order (pure b128 loads in the
// hot loop). bf16 weights (96MB) + hmid (64MB) are L2-resident (192MB L2).
// A-tiles are staged into double-buffered LDS with GLOBAL_LOAD_ASYNC_TO_LDS
// (ASYNCcnt) so the global->LDS stream of chunk k+1 overlaps WMMA on chunk k.
// ---------------------------------------------------------------------------

#define TOK   8192      // B*S
#define HD    1024      // hidden
#define FD    2048      // ffn
#define NEXP  8
#define WEXP  (2097152) // H*F elements per expert (same for w0/w1/wo)

typedef __attribute__((ext_vector_type(16))) __bf16 v16bf;
typedef __attribute__((ext_vector_type(8)))  float  v8f;
typedef int v4i __attribute__((vector_size(16)));

#define AS1 __attribute__((address_space(1)))
#define AS3 __attribute__((address_space(3)))

union FragU { v16bf v; uint4 q[2]; };

__device__ __forceinline__ v8f wmma_bf16(v16bf a, v16bf b, v8f c) {
  // (neg_a, A, neg_b, B, c_mod, C, reuse_a, reuse_b)
  return __builtin_amdgcn_wmma_f32_16x16x32_bf16(false, a, false, b, (short)0, c,
                                                 false, false);
}

// Async 16B copy global -> LDS (no VGPR staging, tracked by ASYNCcnt).
// LDS generic pointers carry the LDS offset in the low 32 bits (aperture in
// the high bits), so truncation gives the 32-bit addrspace(3) address.
__device__ __forceinline__ void async_cp16(const void* g, void* l) {
  __builtin_amdgcn_global_load_async_to_lds_b128(
      (AS1 v4i*)(size_t)g,
      (AS3 v4i*)(unsigned)(size_t)l, 0, 0);
}
__device__ __forceinline__ void wait_async0() {
  asm volatile("s_wait_asynccnt 0x0" ::: "memory");
}

// B fragment (32x16, KxN): lane L holds column N=L%16; elems 0..7 = K half
// (0..7 or 8..15 by lane half), elems 8..15 = same +16. Packed layout stores
// each lane's 16 bf16 contiguously -> 32B per lane per fragment.
__device__ __forceinline__ v16bf load_b_frag(const __bf16* __restrict__ pack,
                                             int frag, int lane) {
  const __bf16* p = pack + ((size_t)frag << 9) + (lane << 4);
  FragU f;
  f.q[0] = *reinterpret_cast<const uint4*>(p);
  f.q[1] = *reinterpret_cast<const uint4*>(p + 8);
  return f.v;
}

// A fragment (16x32, MxK) from an LDS chunk laid out [128 rows][64 cols] bf16.
// lane: M = m_tile*16 + lane%16 ; elems 0..7 -> K = c0+kh..c0+kh+7,
// elems 8..15 -> +16. Both halves are 16B-contiguous in a row-major row.
__device__ __forceinline__ v16bf load_a_frag(const __bf16* sX, int m_tile,
                                             int lane, int c0) {
  int M  = m_tile * 16 + (lane & 15);
  int kh = (lane < 16) ? 0 : 8;
  const __bf16* p = sX + M * 64 + c0 + kh;
  FragU f;
  f.q[0] = *reinterpret_cast<const uint4*>(p);
  f.q[1] = *reinterpret_cast<const uint4*>(p + 16);
  return f.v;
}

// ---------------------------------------------------------------------------
__global__ void moe_zero_cnt(int* cnt) {
  if (threadIdx.x < NEXP) cnt[threadIdx.x] = 0;
}

__global__ void moe_cvt_x(const float* __restrict__ x, __bf16* __restrict__ xb,
                          int n) {
  for (int i = blockIdx.x * blockDim.x + threadIdx.x; i < n;
       i += gridDim.x * blockDim.x)
    xb[i] = (__bf16)x[i];
}

// Pack w0/w1/wo (fp32, [E,K,N]) into per-fragment bf16 layout.
// One block packs one 32x16 fragment (512 elements, 2 per thread).
// frag id within expert: fi = n_tile*K_TILES + k_tile  (k innermost -> the
// GEMM k-loop streams contiguously).
__global__ __launch_bounds__(256) void moe_pack_w(
    const float* __restrict__ w0, const float* __restrict__ w1,
    const float* __restrict__ wo, __bf16* __restrict__ p0,
    __bf16* __restrict__ p1, __bf16* __restrict__ po) {
  int gid = blockIdx.x;                 // 0..98303
  int mat = gid / 32768;                // 0:w0 1:w1 2:wo
  int fr  = gid % 32768;
  int ex  = fr / 4096;
  int fi  = fr % 4096;

  const float* W; __bf16* P; int Ndim, Ktiles;
  if (mat == 0)      { W = w0; P = p0; Ndim = FD; Ktiles = HD / 32; }
  else if (mat == 1) { W = w1; P = p1; Ndim = FD; Ktiles = HD / 32; }
  else               { W = wo; P = po; Ndim = HD; Ktiles = FD / 32; }

  int n_tile = fi / Ktiles;
  int k_tile = fi % Ktiles;
  size_t eb = (size_t)ex * WEXP;

#pragma unroll
  for (int j = 0; j < 2; ++j) {
    int idx  = threadIdx.x * 2 + j;     // 0..511
    int lane = idx >> 4;
    int e    = idx & 15;
    int N    = n_tile * 16 + (lane & 15);
    int kh   = (lane < 16) ? 0 : 8;
    int K    = k_tile * 32 + ((e < 8) ? (kh + e) : (16 + kh + (e - 8)));
    P[eb + (size_t)fi * 512 + idx] = (__bf16)W[eb + (size_t)K * Ndim + N];
  }
}

// Router: one wave32 per token. Lane-parallel dot products over H, butterfly
// reduce, top-2 + softmax on lane 0, atomic append to per-expert lists.
__global__ __launch_bounds__(256) void moe_router(
    const float* __restrict__ x, const float* __restrict__ wr, int* cnt,
    int* __restrict__ list_tok, int* __restrict__ list_slot,
    float* __restrict__ slot_w) {
  int lane  = threadIdx.x & 31;
  int wave  = threadIdx.x >> 5;
  int token = blockIdx.x * 8 + wave;

  float acc[NEXP];
#pragma unroll
  for (int e = 0; e < NEXP; ++e) acc[e] = 0.f;

  const float* xp = x + (size_t)token * HD;
  for (int h = lane; h < HD; h += 32) {
    float xv = xp[h];
    const float* w = wr + h * NEXP;
#pragma unroll
    for (int e = 0; e < NEXP; ++e) acc[e] += xv * w[e];
  }
#pragma unroll
  for (int e = 0; e < NEXP; ++e)
#pragma unroll
    for (int off = 16; off > 0; off >>= 1)
      acc[e] += __shfl_xor(acc[e], off, 32);

  if (lane == 0) {
    int i0 = 0; float v0 = acc[0];
#pragma unroll
    for (int e = 1; e < NEXP; ++e)
      if (acc[e] > v0) { v0 = acc[e]; i0 = e; }
    int i1 = -1; float v1 = -3.4e38f;
#pragma unroll
    for (int e = 0; e < NEXP; ++e)
      if (e != i0 && acc[e] > v1) { v1 = acc[e]; i1 = e; }
    float e1 = __expf(v1 - v0);        // stable softmax over {v0, v1}
    float inv = 1.f / (1.f + e1);
    float p0 = inv, p1 = e1 * inv;

    int pos0 = atomicAdd(&cnt[i0], 1);
    list_tok[i0 * TOK + pos0]  = token;
    list_slot[i0 * TOK + pos0] = token * 2;
    slot_w[token * 2]          = p0;

    int pos1 = atomicAdd(&cnt[i1], 1);
    list_tok[i1 * TOK + pos1]  = token;
    list_slot[i1 * TOK + pos1] = token * 2 + 1;
    slot_w[token * 2 + 1]      = p1;
  }
}

// Stage 1: per expert, 128-token tiles. hmid[slot] = gate_w * silu(x@w0)*(x@w1)
// 8 waves: wave w owns n_tiles {w, 8+w, ...}; 4 M-tiles of accumulators per
// pass so each B fragment feeds 4 WMMAs per matrix. A-chunks double-buffered
// in LDS via async global->LDS copies.
__global__ __launch_bounds__(256) void moe_gate_up(
    const __bf16* __restrict__ xb, const __bf16* __restrict__ pw0,
    const __bf16* __restrict__ pw1, const int* __restrict__ cnt,
    const int* __restrict__ list_tok, const int* __restrict__ list_slot,
    const float* __restrict__ slot_w, __bf16* __restrict__ hmid) {
  const int e = blockIdx.y;
  const int n_tok = cnt[e];
  const int row0 = blockIdx.x * 128;
  if (row0 >= n_tok) return;

  __shared__ __bf16 sX[2][128 * 64];
  __shared__ int   sTok[128];
  __shared__ int   sSlot[128];
  __shared__ float sWt[128];

  const int tid = threadIdx.x;
  if (tid < 128) {
    int r = row0 + tid;
    bool v = r < n_tok;
    int li = e * TOK + (v ? r : row0);
    sTok[tid]  = list_tok[li];
    int sl     = list_slot[li];
    sSlot[tid] = sl;
    sWt[tid]   = v ? slot_w[sl] : 0.f;
  }
  __syncthreads();

  const int lane = tid & 31, wave = tid >> 5;
  const __bf16* b0base = pw0 + (size_t)e * WEXP;
  const __bf16* b1base = pw1 + (size_t)e * WEXP;
  const v8f vzero = {0.f, 0.f, 0.f, 0.f, 0.f, 0.f, 0.f, 0.f};

  // Issue async copies for one 128x64 chunk into LDS buffer `buf`.
  auto issue_chunk = [&](int kc, int buf) {
#pragma unroll
    for (int j = 0; j < 4; ++j) {
      int unit = tid + 256 * j;               // 1024 x 16B units
      int r = unit >> 3, qc = unit & 7;
      async_cp16(xb + (size_t)sTok[r] * HD + kc * 64 + qc * 8,
                 &sX[buf][unit * 8]);
    }
  };

  for (int n_iter = 0; n_iter < 16; ++n_iter) {
    const int n_tile = n_iter * 8 + wave;          // 0..127
    for (int mpass = 0; mpass < 2; ++mpass) {
      v8f accG[4], accU[4];
#pragma unroll
      for (int m = 0; m < 4; ++m) { accG[m] = vzero; accU[m] = vzero; }

      issue_chunk(0, 0);
      for (int kc = 0; kc < 16; ++kc) {            // H in chunks of 64
        const int cur = kc & 1;
        wait_async0();          // this wave's copies into buf[cur] landed
        __syncthreads();        // all waves' copies landed; prev compute done
        if (kc + 1 < 16) issue_chunk(kc + 1, 1 - cur);
#pragma unroll
        for (int kt = 0; kt < 2; ++kt) {
          int k_tile = kc * 2 + kt;
          int fb = n_tile * 32 + k_tile;           // Ktiles(H)=32
          v16bf bg = load_b_frag(b0base, fb, lane);
          v16bf bu = load_b_frag(b1base, fb, lane);
#pragma unroll
          for (int m = 0; m < 4; ++m) {
            v16bf a = load_a_frag(sX[cur], mpass * 4 + m, lane, kt * 32);
            accG[m] = wmma_bf16(a, bg, accG[m]);
            accU[m] = wmma_bf16(a, bu, accU[m]);
          }
        }
      }
      // Epilogue: silu(g)*u * gate_weight -> bf16 hmid[slot, n]
      int nbase = n_tile * 16 + (lane & 15);
      int moff  = (lane < 16) ? 0 : 8;
#pragma unroll
      for (int m = 0; m < 4; ++m) {
        int Mb = (mpass * 4 + m) * 16 + moff;
#pragma unroll
        for (int r = 0; r < 8; ++r) {
          int M = Mb + r;
          if (row0 + M < n_tok) {
            float g = accG[m][r], u = accU[m][r];
            float h = (g / (1.f + __expf(-g))) * u * sWt[M];
            hmid[(size_t)sSlot[M] * FD + nbase] = (__bf16)h;
          }
        }
      }
      __syncthreads();   // all reads of sX done before next pass refills buf0
    }
  }
}

// Stage 2: out_part[slot] = hmid[slot] @ wo[e].  N (=H) split into 4 chunks of
// 256 cols via blockIdx.z; K = F = 2048. Same async double-buffered A-chunks.
__global__ __launch_bounds__(256) void moe_down(
    const __bf16* __restrict__ hmid, const __bf16* __restrict__ pwo,
    const int* __restrict__ cnt, const int* __restrict__ list_slot,
    float* __restrict__ out_part) {
  const int e = blockIdx.y;
  const int n_tok = cnt[e];
  const int row0 = blockIdx.x * 128;
  if (row0 >= n_tok) return;
  const int zc = blockIdx.z;                        // 0..3

  __shared__ __bf16 sH[2][128 * 64];
  __shared__ int sSlot[128];

  const int tid = threadIdx.x;
  if (tid < 128) {
    int r = row0 + tid;
    bool v = r < n_tok;
    sSlot[tid] = list_slot[e * TOK + (v ? r : row0)];
  }
  __syncthreads();

  const int lane = tid & 31, wave = tid >> 5;
  const __bf16* bbase = pwo + (size_t)e * WEXP;
  const v8f vzero = {0.f, 0.f, 0.f, 0.f, 0.f, 0.f, 0.f, 0.f};

  auto issue_chunk = [&](int kc, int buf) {
#pragma unroll
    for (int j = 0; j < 4; ++j) {
      int unit = tid + 256 * j;
      int r = unit >> 3, qc = unit & 7;
      async_cp16(hmid + (size_t)sSlot[r] * FD + kc * 64 + qc * 8,
                 &sH[buf][unit * 8]);
    }
  };

  for (int n_iter = 0; n_iter < 2; ++n_iter) {
    const int n_tile = zc * 16 + n_iter * 8 + wave;  // 0..63
    for (int mpass = 0; mpass < 2; ++mpass) {
      v8f acc[4];
#pragma unroll
      for (int m = 0; m < 4; ++m) acc[m] = vzero;

      issue_chunk(0, 0);
      for (int kc = 0; kc < 32; ++kc) {              // F in chunks of 64
        const int cur = kc & 1;
        wait_async0();
        __syncthreads();
        if (kc + 1 < 32) issue_chunk(kc + 1, 1 - cur);
#pragma unroll
        for (int kt = 0; kt < 2; ++kt) {
          int k_tile = kc * 2 + kt;
          int fb = n_tile * 64 + k_tile;             // Ktiles(F)=64
          v16bf b = load_b_frag(bbase, fb, lane);
#pragma unroll
          for (int m = 0; m < 4; ++m) {
            v16bf a = load_a_frag(sH[cur], mpass * 4 + m, lane, kt * 32);
            acc[m] = wmma_bf16(a, b, acc[m]);
          }
        }
      }
      int nbase = n_tile * 16 + (lane & 15);
      int moff  = (lane < 16) ? 0 : 8;
#pragma unroll
      for (int m = 0; m < 4; ++m) {
        int Mb = (mpass * 4 + m) * 16 + moff;
#pragma unroll
        for (int r = 0; r < 8; ++r) {
          int M = Mb + r;
          if (row0 + M < n_tok)
            out_part[(size_t)sSlot[M] * HD + nbase] = acc[m][r];
        }
      }
      __syncthreads();
    }
  }
}

// out[t,h] = part[2t,h] + part[2t+1,h]   (deterministic, no FP atomics)
__global__ void moe_combine(const float* __restrict__ part,
                            float* __restrict__ out, int n) {
  for (int i = blockIdx.x * blockDim.x + threadIdx.x; i < n;
       i += gridDim.x * blockDim.x) {
    int base = i + (i & ~(HD - 1));   // = 2*t*H + h
    out[i] = part[base] + part[base + HD];
  }
}

// ---------------------------------------------------------------------------
// Workspace layout (bytes). Total ~252 MB.
static constexpr size_t OFF_XB   = 0;                                   // 16MB
static constexpr size_t OFF_PW0  = OFF_XB   + (size_t)TOK * HD * 2;
static constexpr size_t OFF_PW1  = OFF_PW0  + (size_t)WEXP * NEXP * 2;  // 32MB
static constexpr size_t OFF_PWO  = OFF_PW1  + (size_t)WEXP * NEXP * 2;
static constexpr size_t OFF_HMID = OFF_PWO  + (size_t)WEXP * NEXP * 2;  // 64MB
static constexpr size_t OFF_OUTP = OFF_HMID + (size_t)2 * TOK * FD * 2; // 64MB
static constexpr size_t OFF_CNT  = OFF_OUTP + (size_t)2 * TOK * HD * 4;
static constexpr size_t OFF_LTOK = OFF_CNT  + 256;
static constexpr size_t OFF_LSLT = OFF_LTOK + (size_t)NEXP * TOK * 4;
static constexpr size_t OFF_SW   = OFF_LSLT + (size_t)NEXP * TOK * 4;

extern "C" void kernel_launch(void* const* d_in, const int* in_sizes, int n_in,
                              void* d_out, int out_size, void* d_ws,
                              size_t ws_size, hipStream_t stream) {
  const float* x  = (const float*)d_in[0];   // [B,S,H]
  const float* wr = (const float*)d_in[1];   // [H,E]
  const float* w0 = (const float*)d_in[2];   // [E,H,F]
  const float* w1 = (const float*)d_in[3];   // [E,H,F]
  const float* wo = (const float*)d_in[4];   // [E,F,H]
  float* out = (float*)d_out;

  char* ws = (char*)d_ws;
  __bf16* xb   = (__bf16*)(ws + OFF_XB);
  __bf16* pw0  = (__bf16*)(ws + OFF_PW0);
  __bf16* pw1  = (__bf16*)(ws + OFF_PW1);
  __bf16* pwo  = (__bf16*)(ws + OFF_PWO);
  __bf16* hmid = (__bf16*)(ws + OFF_HMID);
  float*  outp = (float*)(ws + OFF_OUTP);
  int*    cnt  = (int*)(ws + OFF_CNT);
  int*    ltok = (int*)(ws + OFF_LTOK);
  int*    lslt = (int*)(ws + OFF_LSLT);
  float*  slw  = (float*)(ws + OFF_SW);

  const int nElems = TOK * HD;

  moe_zero_cnt<<<1, 32, 0, stream>>>(cnt);
  moe_cvt_x<<<4096, 256, 0, stream>>>(x, xb, nElems);
  moe_pack_w<<<98304, 256, 0, stream>>>(w0, w1, wo, pw0, pw1, pwo);
  moe_router<<<TOK / 8, 256, 0, stream>>>(x, wr, cnt, ltok, lslt, slw);
  moe_gate_up<<<dim3(TOK / 128, NEXP, 1), 256, 0, stream>>>(
      xb, pw0, pw1, cnt, ltok, lslt, slw, hmid);
  moe_down<<<dim3(TOK / 128, NEXP, 4), 256, 0, stream>>>(hmid, pwo, cnt, lslt,
                                                         outp);
  moe_combine<<<4096, 256, 0, stream>>>(outp, out, nElems);
}
